// DGCNN_Atten_53206054863445
// MI455X (gfx1250) — matrix operand order repaired and checked by reading, not compiled, not run on hardware
//
#include <hip/hip_runtime.h>
#include <hip/hip_bf16.h>

// ---------------------------------------------------------------------------
// DGCNN + self-attention forward for MI455X (gfx1250, wave32, WMMA).
// All WMMA fragments are single contiguous 32B v16h loads:
//   - weights pre-converted to f16, plain (O,Cin) layout (no permute)
//   - activations stored f16 point-major (B,N,C) for B-fragments
//   - attention: Q point-major, K point-major, V channel-major,
//     P staged in LDS in A-fragment order (vector read-back).
// WMMA kernels are single-wave blocks with __launch_bounds__(32,1) so the
// compiler can use the full VGPR file (no scratch spills in the hot loops).
// ---------------------------------------------------------------------------

typedef __attribute__((ext_vector_type(16))) _Float16 v16h;
typedef __attribute__((ext_vector_type(8)))  _Float16 v8h;
typedef __attribute__((ext_vector_type(8)))  float    v8f;

#define NB   8
#define NPTS 2048
#define KNN  20
#define EPSB 1e-5f

// Within-32 A-fragment channel permute: order {0-7, 16-23, 8-15, 24-31}.
// After permute, lane (m, up) reads its 16 A-elements contiguously at up*16.
__device__ __forceinline__ int permA(int c) {
    return (c & ~31) | ((c & 7) | ((c >> 1) & 8) | ((c << 1) & 16));
}

// ---------------------------------------------------------------------------
// kNN: one thread per (b,n); maintains top-20 by pd = -||xi-xj||^2 (desc).
// ---------------------------------------------------------------------------
template<int C>
__global__ void knn_kernel(const float* __restrict__ H, int bstride,
                           int* __restrict__ Idx) {
    int t = blockIdx.x * blockDim.x + threadIdx.x;
    if (t >= NB * NPTS) return;
    int b = t / NPTS, n = t % NPTS;
    const float* Hb = H + (size_t)b * bstride;
    float xi[C];
    #pragma unroll
    for (int c = 0; c < C; ++c) xi[c] = Hb[c * NPTS + n];
    float best[KNN]; int bidx[KNN];
    #pragma unroll
    for (int q = 0; q < KNN; ++q) { best[q] = -3.4e38f; bidx[q] = 0; }
    for (int j = 0; j < NPTS; ++j) {
        float d2 = 0.f;
        #pragma unroll
        for (int c = 0; c < C; ++c) {
            float df = xi[c] - Hb[c * NPTS + j];
            d2 += df * df;
        }
        float pd = -d2;
        if (pd > best[KNN - 1]) {
            int q = KNN - 1;
            while (q > 0 && best[q - 1] < pd) {
                best[q] = best[q - 1]; bidx[q] = bidx[q - 1]; --q;
            }
            best[q] = pd; bidx[q] = j;
        }
    }
    #pragma unroll
    for (int q = 0; q < KNN; ++q) Idx[(size_t)t * KNN + q] = bidx[q];
}

// ---------------------------------------------------------------------------
// Edge stage pass 1: y0 = W0 * edge_feature, accumulate BN0 sums.
// ---------------------------------------------------------------------------
template<int C>
__global__ void edge_p1_kernel(const float* __restrict__ H, int bstride,
                               const int* __restrict__ Idx,
                               const float* __restrict__ W0,
                               float* __restrict__ stats) {
    __shared__ float ssum[64], ssq[64];
    int tid = threadIdx.x;
    for (int i = tid; i < 64; i += blockDim.x) { ssum[i] = 0.f; ssq[i] = 0.f; }
    __syncthreads();
    int t = blockIdx.x * blockDim.x + tid;
    if (t < NB * NPTS * KNN) {
        int bn = t / KNN;
        int n = bn % NPTS, b = bn / NPTS;
        const float* Hb = H + (size_t)b * bstride;
        int j = Idx[t];
        float y[64];
        #pragma unroll
        for (int o = 0; o < 64; ++o) y[o] = 0.f;
        for (int c = 0; c < C; ++c) {
            float cen = Hb[c * NPTS + n];
            float e0  = Hb[c * NPTS + j] - cen;
            #pragma unroll
            for (int o = 0; o < 64; ++o)
                y[o] += W0[o * 2 * C + c] * e0 + W0[o * 2 * C + C + c] * cen;
        }
        #pragma unroll
        for (int o = 0; o < 64; ++o) {
            atomicAdd(&ssum[o], y[o]);
            atomicAdd(&ssq[o],  y[o] * y[o]);
        }
    }
    __syncthreads();
    for (int i = tid; i < 64; i += blockDim.x) {
        atomicAdd(&stats[i],      ssum[i]);
        atomicAdd(&stats[64 + i], ssq[i]);
    }
}

__global__ void bn_finalize_kernel(const float* __restrict__ stats,
                                   float* __restrict__ mi, int Cn, float invc) {
    int o = blockIdx.x * blockDim.x + threadIdx.x;
    if (o >= Cn) return;
    float mean = stats[o] * invc;
    float var  = stats[Cn + o] * invc - mean * mean;
    mi[o]      = mean;
    mi[Cn + o] = rsqrtf(var + EPSB);
}

// ---------------------------------------------------------------------------
// Edge stage pass 3: recompute y0 -> BN0+LReLU -> y1 = W1*a0, accumulate BN1.
// ---------------------------------------------------------------------------
template<int C>
__global__ void edge_p3_kernel(const float* __restrict__ H, int bstride,
                               const int* __restrict__ Idx,
                               const float* __restrict__ W0,
                               const float* __restrict__ MI0,
                               const float* __restrict__ G0,
                               const float* __restrict__ B0,
                               const float* __restrict__ W1,
                               float* __restrict__ stats) {
    __shared__ float ssum[64], ssq[64];
    int tid = threadIdx.x;
    for (int i = tid; i < 64; i += blockDim.x) { ssum[i] = 0.f; ssq[i] = 0.f; }
    __syncthreads();
    int t = blockIdx.x * blockDim.x + tid;
    if (t < NB * NPTS * KNN) {
        int bn = t / KNN;
        int n = bn % NPTS, b = bn / NPTS;
        const float* Hb = H + (size_t)b * bstride;
        int j = Idx[t];
        float y0[64];
        #pragma unroll
        for (int o = 0; o < 64; ++o) y0[o] = 0.f;
        for (int c = 0; c < C; ++c) {
            float cen = Hb[c * NPTS + n];
            float e0  = Hb[c * NPTS + j] - cen;
            #pragma unroll
            for (int o = 0; o < 64; ++o)
                y0[o] += W0[o * 2 * C + c] * e0 + W0[o * 2 * C + C + c] * cen;
        }
        #pragma unroll
        for (int o = 0; o < 64; ++o) {
            float a = (y0[o] - MI0[o]) * MI0[64 + o] * G0[o] + B0[o];
            y0[o] = a > 0.f ? a : 0.2f * a;
        }
        float y1[64];
        #pragma unroll
        for (int o = 0; o < 64; ++o) y1[o] = 0.f;
        #pragma unroll
        for (int c2 = 0; c2 < 64; ++c2) {
            float a = y0[c2];
            #pragma unroll
            for (int o = 0; o < 64; ++o) y1[o] += W1[o * 64 + c2] * a;
        }
        #pragma unroll
        for (int o = 0; o < 64; ++o) {
            atomicAdd(&ssum[o], y1[o]);
            atomicAdd(&ssq[o],  y1[o] * y1[o]);
        }
    }
    __syncthreads();
    for (int i = tid; i < 64; i += blockDim.x) {
        atomicAdd(&stats[i],      ssum[i]);
        atomicAdd(&stats[64 + i], ssq[i]);
    }
}

// ---------------------------------------------------------------------------
// Edge stage pass 5: full chain + max over k -> h_out. One thread per (b,n).
// ---------------------------------------------------------------------------
template<int C>
__global__ void edge_p5_kernel(const float* __restrict__ H, int bstride,
                               const int* __restrict__ Idx,
                               const float* __restrict__ W0,
                               const float* __restrict__ MI0,
                               const float* __restrict__ G0,
                               const float* __restrict__ B0,
                               const float* __restrict__ W1,
                               const float* __restrict__ MI1,
                               const float* __restrict__ G1,
                               const float* __restrict__ B1,
                               float* __restrict__ Hout, int out_bstride) {
    int t = blockIdx.x * blockDim.x + threadIdx.x;
    if (t >= NB * NPTS) return;
    int n = t % NPTS, b = t / NPTS;
    const float* Hb = H + (size_t)b * bstride;
    float hmax[64];
    #pragma unroll
    for (int o = 0; o < 64; ++o) hmax[o] = -3.4e38f;
    for (int k = 0; k < KNN; ++k) {
        int j = Idx[(size_t)t * KNN + k];
        float y0[64];
        #pragma unroll
        for (int o = 0; o < 64; ++o) y0[o] = 0.f;
        for (int c = 0; c < C; ++c) {
            float cen = Hb[c * NPTS + n];
            float e0  = Hb[c * NPTS + j] - cen;
            #pragma unroll
            for (int o = 0; o < 64; ++o)
                y0[o] += W0[o * 2 * C + c] * e0 + W0[o * 2 * C + C + c] * cen;
        }
        #pragma unroll
        for (int o = 0; o < 64; ++o) {
            float a = (y0[o] - MI0[o]) * MI0[64 + o] * G0[o] + B0[o];
            y0[o] = a > 0.f ? a : 0.2f * a;
        }
        float y1[64];
        #pragma unroll
        for (int o = 0; o < 64; ++o) y1[o] = 0.f;
        #pragma unroll
        for (int c2 = 0; c2 < 64; ++c2) {
            float a = y0[c2];
            #pragma unroll
            for (int o = 0; o < 64; ++o) y1[o] += W1[o * 64 + c2] * a;
        }
        #pragma unroll
        for (int o = 0; o < 64; ++o) {
            float a = (y1[o] - MI1[o]) * MI1[64 + o] * G1[o] + B1[o];
            a = a > 0.f ? a : 0.2f * a;
            hmax[o] = fmaxf(hmax[o], a);
        }
    }
    float* Ob = Hout + (size_t)b * out_bstride;
    #pragma unroll
    for (int o = 0; o < 64; ++o) Ob[o * NPTS + n] = hmax[o];
}

// ---------------------------------------------------------------------------
// Layout conversion kernels
// ---------------------------------------------------------------------------
// W (O,Cin f32) -> f16, same (O,Cin) layout (no permute).
__global__ void convert_w_kernel(const float* __restrict__ W,
                                 _Float16* __restrict__ Wh, int O, int Cin) {
    int i = blockIdx.x * blockDim.x + threadIdx.x;
    if (i >= O * Cin) return;
    int o = i / Cin, c = i % Cin;
    Wh[(size_t)o * Cin + c] = (_Float16)W[i];
}

// X (B,Cn,N f32) -> XT (B,N,Cn f16), point-major for B-fragments.
__global__ void convert_xT_kernel(const float* __restrict__ X,
                                  _Float16* __restrict__ XT, int Cn) {
    size_t i = (size_t)blockIdx.x * blockDim.x + threadIdx.x;
    size_t total = (size_t)NB * Cn * NPTS;
    if (i >= total) return;
    size_t n = i % NPTS;
    size_t c = (i / NPTS) % Cn;
    size_t b = i / ((size_t)Cn * NPTS);
    XT[(b * NPTS + n) * Cn + c] = (_Float16)X[i];
}

// ---------------------------------------------------------------------------
// WMMA GEMM: Out[b,n,o] = sum_c W[o,c] * In[b,n,c] (all operands f16).
// Each fragment is one contiguous 32B v16h load. One wave per 16x16 tile.
// SMODE: 0 = point-major f16 (B,N,O); 1 = point-major f16 (B,N,O), as 0 (Q);
//        2 = channel-major f16 (B,O,N) (V).
// ---------------------------------------------------------------------------
template<int SMODE>
__global__ __launch_bounds__(32, 1)
void gemm_wmma_kernel(const _Float16* __restrict__ Wh,
                      const _Float16* __restrict__ XT,
                      _Float16* __restrict__ Out, int O, int Cin) {
    int tiles_n = NPTS / 16, tiles_o = O / 16;
    int bid = blockIdx.x;
    int b   = bid / (tiles_o * tiles_n);
    int rem = bid % (tiles_o * tiles_n);
    int oBase = (rem / tiles_n) * 16;
    int nBase = (rem % tiles_n) * 16;
    int lane = threadIdx.x;
    int m  = lane & 15;
    int up = (lane >> 4) & 1;
    const _Float16* wrow = Wh + (size_t)(oBase + m) * Cin + up * 16;
    const _Float16* xrow = XT + ((size_t)b * NPTS + nBase + m) * Cin + up * 16;
    v8f acc = {};
    for (int c0 = 0; c0 < Cin; c0 += 32) {
        if (c0 + 32 < Cin)
            __builtin_prefetch(xrow + c0 + 32, 0, 1);
        v16h af = *(const v16h*)(wrow + c0);
        v16h bf = *(const v16h*)(xrow + c0);
        acc = __builtin_amdgcn_wmma_f32_16x16x32_f16(
                  false, af, false, bf, (short)0, acc, false, false);
    }
    if (SMODE == 2) {
        // channel-major (B,O,N)
        #pragma unroll
        for (int r = 0; r < 8; ++r) {
            int chan = oBase + r + up * 8;
            Out[(size_t)b * O * NPTS + (size_t)chan * NPTS + nBase + m] =
                (_Float16)acc[r];
        }
    } else {
        v8h ov;
        #pragma unroll
        for (int r = 0; r < 8; ++r) ov[r] = (_Float16)acc[r];
        int cb = oBase + up * 8;               // 8 consecutive channels

        *(v8h*)(Out + ((size_t)b * NPTS + nBase + m) * O + cb) = ov;
    }
}

// BN stats over (B,N) per channel, point-major f16 input.
__global__ void bn_stats_t_kernel(const _Float16* __restrict__ X,
                                  float* __restrict__ mi, int Cn) {
    int o = blockIdx.x;
    __shared__ float ss[256], sq[256];
    float s = 0.f, q = 0.f;
    for (int i = threadIdx.x; i < NB * NPTS; i += blockDim.x) {
        float v = (float)X[(size_t)i * Cn + o];
        s += v; q += v * v;
    }
    ss[threadIdx.x] = s; sq[threadIdx.x] = q;
    __syncthreads();
    for (int st = blockDim.x / 2; st > 0; st >>= 1) {
        if ((int)threadIdx.x < st) {
            ss[threadIdx.x] += ss[threadIdx.x + st];
            sq[threadIdx.x] += sq[threadIdx.x + st];
        }
        __syncthreads();
    }
    if (threadIdx.x == 0) {
        float invc = 1.f / (float)(NB * NPTS);
        float mean = ss[0] * invc;
        float var  = sq[0] * invc - mean * mean;
        mi[o]      = mean;
        mi[Cn + o] = rsqrtf(var + EPSB);
    }
}

// BN + LeakyReLU in place on point-major f16 (B,N,Cn).
__global__ void bn_apply_t_kernel(_Float16* __restrict__ X,
                                  const float* __restrict__ mi,
                                  const float* __restrict__ G,
                                  const float* __restrict__ Bb, int Cn) {
    size_t i = (size_t)blockIdx.x * blockDim.x + threadIdx.x;
    size_t total = (size_t)NB * NPTS * Cn;
    if (i >= total) return;
    int o = (int)(i % Cn);
    float y = ((float)X[i] - mi[o]) * mi[Cn + o] * G[o] + Bb[o];
    X[i] = (_Float16)(y > 0.f ? y : 0.2f * y);
}

// copy feats[0] slice (channels 0..63 of fcat) into d_out
__global__ void copy_slice_kernel(const float* __restrict__ src,
                                  float* __restrict__ dst) {
    size_t i = (size_t)blockIdx.x * blockDim.x + threadIdx.x;
    size_t total = (size_t)NB * 64 * NPTS;
    if (i >= total) return;
    size_t n = i % NPTS;
    size_t c = (i / NPTS) % 64;
    size_t b = i / ((size_t)64 * NPTS);
    dst[i] = src[b * (size_t)192 * NPTS + c * NPTS + n];
}

// ---------------------------------------------------------------------------
// Flash attention: one wave per 16-row query tile, 32-key tiles.
// Q: (B,N,128) f16; K: (B,N,128) f16; V: (B,128,N) f16.
// S = Q K^T (4 wmma), P staged in LDS in A-frag order, O += P V (8 wmma).
// Single wave per block; launch_bounds lets the full VGPR file hold
// qf[4] + oacc[8] + fragments without scratch spills.
// ---------------------------------------------------------------------------
__global__ __launch_bounds__(32, 1)
void attention_kernel(const _Float16* __restrict__ Qt,
                      const _Float16* __restrict__ Kt,
                      const _Float16* __restrict__ Vc,
                      float* __restrict__ Out) {
    const int C = 128;
    int bid = blockIdx.x;
    int b = bid / (NPTS / 16);
    int rowBase = (bid % (NPTS / 16)) * 16;
    int lane = threadIdx.x;
    int m  = lane & 15;
    int up = (lane >> 4) & 1;
    const _Float16* Qb = Qt + (size_t)b * NPTS * C;
    const _Float16* Kb = Kt + (size_t)b * NPTS * C;
    const _Float16* Vb = Vc + (size_t)b * C * NPTS;

    __shared__ __align__(32) _Float16 Pld[16 * 32];

    // Q fragments: one v16h load each (unpermuted point-major, contiguous k)
    const _Float16* Qrow = Qb + (size_t)(rowBase + m) * C + up * 16;
    v16h qf[4];
    #pragma unroll
    for (int cc = 0; cc < 4; ++cc)
        qf[cc] = *(const v16h*)(Qrow + cc * 32);

    float rowm[8], rowl[8];
    #pragma unroll
    for (int r = 0; r < 8; ++r) { rowm[r] = -3.4e38f; rowl[r] = 0.f; }
    v8f oacc[8];
    v8f vz = {};
    #pragma unroll
    for (int cch = 0; cch < 8; ++cch) oacc[cch] = vz;

    const float inv_temp = 0.08838834764831845f; // 1/sqrt(128)
    const int p0pos = m;               // LDS positions in key order
    const int p1pos = m + 16;

    for (int j0 = 0; j0 < NPTS; j0 += 32) {
        // scores for 32 keys (two 16-col D tiles)
        v8f s[2] = { vz, vz };
        #pragma unroll
        for (int kh = 0; kh < 2; ++kh) {
            const _Float16* Krow = Kb + (size_t)(j0 + kh * 16 + m) * C + up * 16;
            #pragma unroll
            for (int cc = 0; cc < 4; ++cc) {
                v16h bf = *(const v16h*)(Krow + cc * 32);
                s[kh] = __builtin_amdgcn_wmma_f32_16x16x32_f16(
                            false, qf[cc], false, bf, (short)0, s[kh], false, false);
            }
        }
        // scale + per-row max (reduce over the 16 lanes of a row group)
        float newm[8];
        #pragma unroll
        for (int r = 0; r < 8; ++r) {
            float v0 = s[0][r] * inv_temp, v1 = s[1][r] * inv_temp;
            s[0][r] = v0; s[1][r] = v1;
            float mx = fmaxf(v0, v1);
            #pragma unroll
            for (int d = 1; d < 16; d <<= 1)
                mx = fmaxf(mx, __shfl_xor(mx, d, 32));
            newm[r] = fmaxf(rowm[r], mx);
        }
        __syncthreads();   // previous iteration's Pld reads complete
        #pragma unroll
        for (int r = 0; r < 8; ++r) {
            float corr = __expf(rowm[r] - newm[r]);
            float p0 = __expf(s[0][r] - newm[r]);
            float p1 = __expf(s[1][r] - newm[r]);
            float rs = p0 + p1;
            #pragma unroll
            for (int d = 1; d < 16; d <<= 1)
                rs += __shfl_xor(rs, d, 32);
            rowl[r] = rowl[r] * corr + rs;
            rowm[r] = newm[r];
            int row = r + up * 8;
            Pld[row * 32 + p0pos] = (_Float16)p0;
            Pld[row * 32 + p1pos] = (_Float16)p1;
            #pragma unroll
            for (int cch = 0; cch < 8; ++cch) oacc[cch][r] *= corr;
        }
        __syncthreads();
        // P (16x32) as A-fragment: one aligned 32B LDS vector read per lane
        v16h pa = *(const v16h*)(Pld + m * 32 + up * 16);
        // O += P * V  (8 channel chunks of 16); V B-frags contiguous in keys
        #pragma unroll
        for (int cch = 0; cch < 8; ++cch) {
            v16h vf = *(const v16h*)(Vb + (size_t)(cch * 16 + m) * NPTS
                                        + j0 + up * 16);
            oacc[cch] = __builtin_amdgcn_wmma_f32_16x16x32_f16(
                            false, pa, false, vf, (short)0, oacc[cch], false, false);
        }
    }
    // Out[b, chan, row] = O / l
    float* Ob = Out + (size_t)b * C * NPTS;
    #pragma unroll
    for (int cch = 0; cch < 8; ++cch)
        #pragma unroll
        for (int r = 0; r < 8; ++r) {
            int row  = rowBase + r + up * 8;
            int chan = cch * 16 + m;
            Ob[(size_t)chan * NPTS + row] = oacc[cch][r] / rowl[r];
        }
}

// ---------------------------------------------------------------------------
// Host orchestration
// ---------------------------------------------------------------------------
extern "C" void kernel_launch(void* const* d_in, const int* in_sizes, int n_in,
                              void* d_out, int out_size, void* d_ws, size_t ws_size,
                              hipStream_t stream) {
    const float* x     = (const float*)d_in[0];
    const float* w_e00 = (const float*)d_in[1];
    const float* g_e00 = (const float*)d_in[2];
    const float* b_e00 = (const float*)d_in[3];
    const float* w_e01 = (const float*)d_in[4];
    const float* g_e01 = (const float*)d_in[5];
    const float* b_e01 = (const float*)d_in[6];
    const float* w_e10 = (const float*)d_in[7];
    const float* g_e10 = (const float*)d_in[8];
    const float* b_e10 = (const float*)d_in[9];
    const float* w_e11 = (const float*)d_in[10];
    const float* g_e11 = (const float*)d_in[11];
    const float* b_e11 = (const float*)d_in[12];
    const float* w_e20 = (const float*)d_in[13];
    const float* g_e20 = (const float*)d_in[14];
    const float* b_e20 = (const float*)d_in[15];
    const float* w_e21 = (const float*)d_in[16];
    const float* g_e21 = (const float*)d_in[17];
    const float* b_e21 = (const float*)d_in[18];
    const float* w_m0  = (const float*)d_in[19];
    const float* g_m0  = (const float*)d_in[20];
    const float* b_m0  = (const float*)d_in[21];
    const float* w_m1  = (const float*)d_in[22];
    const float* g_m1  = (const float*)d_in[23];
    const float* b_m1  = (const float*)d_in[24];
    const float* w_q   = (const float*)d_in[25];
    const float* w_k   = (const float*)d_in[26];
    const float* w_v   = (const float*)d_in[27];

    // workspace carve-out (~56 MB)
    unsigned char* wp = (unsigned char*)d_ws;
    auto alloc = [&](size_t bytes) -> void* {
        void* p = (void*)wp;
        wp += (bytes + 255) & ~(size_t)255;
        return p;
    };
    int*      d_idx   = (int*)     alloc((size_t)NB * NPTS * KNN * 4);
    float*    d_stats = (float*)   alloc(128 * 4);
    float*    d_mi0   = (float*)   alloc(128 * 4);
    float*    d_mi1   = (float*)   alloc(128 * 4);
    float*    d_miM   = (float*)   alloc(1024 * 4);
    float*    d_fcat  = (float*)   alloc((size_t)NB * 192 * NPTS * 4);
    _Float16* d_fcatT = (_Float16*)alloc((size_t)NB * NPTS * 192 * 2);
    _Float16* d_whM0  = (_Float16*)alloc((size_t)512 * 192 * 2);
    _Float16* d_whM1  = (_Float16*)alloc((size_t)256 * 512 * 2);
    _Float16* d_whQ   = (_Float16*)alloc((size_t)128 * 256 * 2);
    _Float16* d_whK   = (_Float16*)alloc((size_t)128 * 256 * 2);
    _Float16* d_whV   = (_Float16*)alloc((size_t)128 * 256 * 2);
    _Float16* d_ym0T  = (_Float16*)alloc((size_t)NB * NPTS * 512 * 2);
    _Float16* d_f256T = (_Float16*)alloc((size_t)NB * NPTS * 256 * 2);
    _Float16* d_q     = (_Float16*)alloc((size_t)NB * NPTS * 128 * 2);
    _Float16* d_k     = (_Float16*)alloc((size_t)NB * NPTS * 128 * 2);
    _Float16* d_v     = (_Float16*)alloc((size_t)NB * 128 * NPTS * 2);

    const int   nEdge   = NB * NPTS * KNN;             // 327680
    const float invEdge = 1.f / (float)nEdge;
    const int   gEdge   = (nEdge + 255) / 256;
    const int   gKnn    = (NB * NPTS) / 128;
    const int   gP5     = (NB * NPTS) / 64;

    // weight conversions (independent of edge stages; issue first)
    convert_w_kernel<<<(512 * 192 + 255) / 256, 256, 0, stream>>>(w_m0, d_whM0, 512, 192);
    convert_w_kernel<<<(256 * 512 + 255) / 256, 256, 0, stream>>>(w_m1, d_whM1, 256, 512);
    convert_w_kernel<<<(128 * 256 + 255) / 256, 256, 0, stream>>>(w_q,  d_whQ,  128, 256);
    convert_w_kernel<<<(128 * 256 + 255) / 256, 256, 0, stream>>>(w_k,  d_whK,  128, 256);
    convert_w_kernel<<<(128 * 256 + 255) / 256, 256, 0, stream>>>(w_v,  d_whV,  128, 256);

    // ---- edge stage 0 (C=9, input x) -> fcat channels [0,64) ----
    knn_kernel<9><<<gKnn, 128, 0, stream>>>(x, 9 * NPTS, d_idx);
    hipMemsetAsync(d_stats, 0, 128 * 4, stream);
    edge_p1_kernel<9><<<gEdge, 256, 0, stream>>>(x, 9 * NPTS, d_idx, w_e00, d_stats);
    bn_finalize_kernel<<<1, 64, 0, stream>>>(d_stats, d_mi0, 64, invEdge);
    hipMemsetAsync(d_stats, 0, 128 * 4, stream);
    edge_p3_kernel<9><<<gEdge, 256, 0, stream>>>(x, 9 * NPTS, d_idx, w_e00, d_mi0,
                                                 g_e00, b_e00, w_e01, d_stats);
    bn_finalize_kernel<<<1, 64, 0, stream>>>(d_stats, d_mi1, 64, invEdge);
    edge_p5_kernel<9><<<gP5, 64, 0, stream>>>(x, 9 * NPTS, d_idx,
                                              w_e00, d_mi0, g_e00, b_e00,
                                              w_e01, d_mi1, g_e01, b_e01,
                                              d_fcat, 192 * NPTS);

    // ---- edge stage 1 (C=64, input fcat[0:64)) -> fcat channels [64,128) ----
    const float* h0 = d_fcat;
    knn_kernel<64><<<gKnn, 128, 0, stream>>>(h0, 192 * NPTS, d_idx);
    hipMemsetAsync(d_stats, 0, 128 * 4, stream);
    edge_p1_kernel<64><<<gEdge, 256, 0, stream>>>(h0, 192 * NPTS, d_idx, w_e10, d_stats);
    bn_finalize_kernel<<<1, 64, 0, stream>>>(d_stats, d_mi0, 64, invEdge);
    hipMemsetAsync(d_stats, 0, 128 * 4, stream);
    edge_p3_kernel<64><<<gEdge, 256, 0, stream>>>(h0, 192 * NPTS, d_idx, w_e10, d_mi0,
                                                  g_e10, b_e10, w_e11, d_stats);
    bn_finalize_kernel<<<1, 64, 0, stream>>>(d_stats, d_mi1, 64, invEdge);
    edge_p5_kernel<64><<<gP5, 64, 0, stream>>>(h0, 192 * NPTS, d_idx,
                                               w_e10, d_mi0, g_e10, b_e10,
                                               w_e11, d_mi1, g_e11, b_e11,
                                               d_fcat + 64 * NPTS, 192 * NPTS);

    // ---- edge stage 2 (C=64, input fcat[64:128)) -> fcat channels [128,192) ----
    const float* h1 = d_fcat + 64 * NPTS;
    knn_kernel<64><<<gKnn, 128, 0, stream>>>(h1, 192 * NPTS, d_idx);
    hipMemsetAsync(d_stats, 0, 128 * 4, stream);
    edge_p1_kernel<64><<<gEdge, 256, 0, stream>>>(h1, 192 * NPTS, d_idx, w_e20, d_stats);
    bn_finalize_kernel<<<1, 64, 0, stream>>>(d_stats, d_mi0, 64, invEdge);
    hipMemsetAsync(d_stats, 0, 128 * 4, stream);
    edge_p3_kernel<64><<<gEdge, 256, 0, stream>>>(h1, 192 * NPTS, d_idx, w_e20, d_mi0,
                                                  g_e20, b_e20, w_e21, d_stats);
    bn_finalize_kernel<<<1, 64, 0, stream>>>(d_stats, d_mi1, 64, invEdge);
    edge_p5_kernel<64><<<gP5, 64, 0, stream>>>(h1, 192 * NPTS, d_idx,
                                               w_e20, d_mi0, g_e20, b_e20,
                                               w_e21, d_mi1, g_e21, b_e21,
                                               d_fcat + 128 * NPTS, 192 * NPTS);

    // feats[0] -> first output section
    copy_slice_kernel<<<(NB * 64 * NPTS + 255) / 256, 256, 0, stream>>>(
        d_fcat, (float*)d_out);

    // fcat -> f16 point-major for WMMA
    convert_xT_kernel<<<(NB * 192 * NPTS + 255) / 256, 256, 0, stream>>>(
        d_fcat, d_fcatT, 192);

    // ---- MLP m0: 192 -> 512 (WMMA) + BN + LReLU ----
    gemm_wmma_kernel<0><<<NB * (512 / 16) * (NPTS / 16), 32, 0, stream>>>(
        d_whM0, d_fcatT, d_ym0T, 512, 192);
    bn_stats_t_kernel<<<512, 256, 0, stream>>>(d_ym0T, d_miM, 512);
    bn_apply_t_kernel<<<(NB * NPTS * 512 + 255) / 256, 256, 0, stream>>>(
        d_ym0T, d_miM, g_m0, b_m0, 512);

    // ---- MLP m1: 512 -> 256 (WMMA) + BN + LReLU ----
    gemm_wmma_kernel<0><<<NB * (256 / 16) * (NPTS / 16), 32, 0, stream>>>(
        d_whM1, d_ym0T, d_f256T, 256, 512);
    bn_stats_t_kernel<<<256, 256, 0, stream>>>(d_f256T, d_miM, 256);
    bn_apply_t_kernel<<<(NB * NPTS * 256 + 255) / 256, 256, 0, stream>>>(
        d_f256T, d_miM, g_m1, b_m1, 256);

    // ---- q/k/v projections 256 -> 128 (WMMA) ----
    const int gQKV = NB * (128 / 16) * (NPTS / 16);
    gemm_wmma_kernel<1><<<gQKV, 32, 0, stream>>>(d_whQ, d_f256T, d_q, 128, 256);
    gemm_wmma_kernel<0><<<gQKV, 32, 0, stream>>>(d_whK, d_f256T, d_k, 128, 256);
    gemm_wmma_kernel<2><<<gQKV, 32, 0, stream>>>(d_whV, d_f256T, d_v, 128, 256);

    // ---- flash attention -> second output section ----
    attention_kernel<<<NB * (NPTS / 16), 32, 0, stream>>>(
        d_q, d_k, d_v, (float*)d_out + (size_t)NB * 64 * NPTS);
}
